// Encoder_4999341932803
// MI455X (gfx1250) — matrix operand order repaired and checked by reading, not compiled, not run on hardware
//
#include <hip/hip_runtime.h>
#include <hip/hip_bf16.h>

typedef __attribute__((ext_vector_type(16))) __bf16 v16bf;
typedef __attribute__((ext_vector_type(8)))  float  v8f;

#define TSTEPS 512
#define BATCH  128
#define IDIM   128
#define HDIM   512
#define KDIM   (IDIM + HDIM)   /* 640 */
#define LDS_STRIDE 648         /* bf16 elems per LDS row, padded (8*81: keeps b128 align, spreads banks) */
#define NWG    64
#define THREADS 128            /* 4 waves of 32 */

__device__ __forceinline__ unsigned short f2bf_sw(float f) {
  unsigned u = __float_as_uint(f);
  u += 0x7FFFu + ((u >> 16) & 1u);           // round-to-nearest-even (software path)
  return (unsigned short)(u >> 16);
}

// Pack two f32 -> two bf16 in one dword via the native CDNA5 instruction (probe-verified).
__device__ __forceinline__ unsigned pk2(float a, float b) {
#if __has_builtin(__builtin_amdgcn_cvt_pk_bf16_f32)
  typedef __attribute__((ext_vector_type(2))) __bf16 v2bf;
  union { v2bf v; unsigned u; } c;
  c.v = __builtin_amdgcn_cvt_pk_bf16_f32(a, b);
  return c.u;
#else
  unsigned r;
  asm("v_cvt_pk_bf16_f32 %0, %1, %2" : "=v"(r) : "v"(a), "v"(b));
  return r;
#endif
}
__device__ __forceinline__ unsigned short f2bf_hw(float f) {
  return (unsigned short)(pk2(f, f) & 0xFFFFu);
}
__device__ __forceinline__ float fsig(float x) {
  return __builtin_amdgcn_rcpf(1.f + __expf(-x));
}
__device__ __forceinline__ float ftanh(float x) {
  return 2.f * __builtin_amdgcn_rcpf(1.f + __expf(-2.f * x)) - 1.f;
}
__device__ __forceinline__ v16bf ld_frag_lds(const unsigned short* p) {
  union { uint4 q[2]; v16bf v; } u;
  u.q[0] = *(const uint4*)p;
  u.q[1] = *(const uint4*)(p + 16);
  return u.v;
}

// ---------------- prep: h0 -> bf16 state buffer 0; reset grid barrier ----------------
__global__ void lstm_prep(const float* __restrict__ h0,
                          unsigned short* __restrict__ hbuf,
                          unsigned* __restrict__ counter) {
  int idx = blockIdx.x * blockDim.x + threadIdx.x;
  if (idx == 0) *counter = 0u;
  if (idx < BATCH * HDIM) hbuf[idx] = f2bf_sw(h0[idx]);
}

// ---------------- persistent LSTM over all T steps ----------------
__global__ __launch_bounds__(THREADS) void lstm_persistent(
    const float* __restrict__ x,     // [B,T,I]
    const float* __restrict__ Wih,   // [4H,I]
    const float* __restrict__ Whh,   // [4H,H]
    const float* __restrict__ bih,
    const float* __restrict__ bhh,
    const float* __restrict__ c0,    // [B,H]
    unsigned short* hbuf,            // [2][B*H] bf16 (double buffered)
    unsigned* counter,
    float* __restrict__ out)         // [B*H last_out][B*T*H input_encoded]
{
  extern __shared__ unsigned short wlds[];   // [64 rows][LDS_STRIDE]

  const int wg    = blockIdx.x;              // 0..63
  const int nb    = wg & 31;                 // hidden block (16 wide)
  const int mhalf = wg >> 5;                 // 0..1
  const int wave  = threadIdx.x >> 5;        // 0..3
  const int lane  = threadIdx.x & 31;
  const int mb    = mhalf * 4 + wave;        // batch block 0..7
  const int m16   = lane & 15;
  const int hi    = lane >> 4;

  // ---- load this WG's weight slice into LDS (bf16), rows = gate*16 + n ----
  for (int e = threadIdx.x; e < 64 * (KDIM / 2); e += THREADS) {
    int row  = e / (KDIM / 2);
    int kp   = (e % (KDIM / 2)) * 2;         // even k
    int gate = row >> 4, n = row & 15;
    int grow = gate * HDIM + nb * 16 + n;    // global output-gate row
    float f0, f1;
    if (kp < IDIM) { f0 = Wih[grow * IDIM + kp];          f1 = Wih[grow * IDIM + kp + 1]; }
    else           { f0 = Whh[grow * HDIM + (kp - IDIM)]; f1 = Whh[grow * HDIM + (kp - IDIM) + 1]; }
    *(unsigned*)&wlds[row * LDS_STRIDE + kp] = pk2(f0, f1);
  }
  __syncthreads();

  // ---- per-lane constants ----
  const int hg = nb * 16 + m16;              // global hidden index of C/D column
  v8f vbias[4];                              // bias lives in the WMMA C operand of the first x-chunk
#pragma unroll
  for (int g = 0; g < 4; ++g) {
    float bv = bih[g * HDIM + hg] + bhh[g * HDIM + hg];
#pragma unroll
    for (int r = 0; r < 8; ++r) vbias[g][r] = bv;
  }

  const int b_base = mb * 16 + hi * 8;       // batch of C/D element r: b_base + r
  float creg[8];
#pragma unroll
  for (int r = 0; r < 8; ++r) creg[r] = c0[(b_base + r) * HDIM + hg];

  const int arow = mb * 16 + m16;            // A-fragment row (batch) for this lane
  const unsigned short* ldsb = &wlds[m16 * LDS_STRIDE + hi * 8];  // per-lane LDS base
  const float* xbase = x + (size_t)arow * TSTEPS * IDIM;

  // x-partial for timestep tt, seeded with the bias (C operand of chunk 0) -> acc[0..3]
  auto compute_x_part = [&](int tt, v8f* acc) {
    const float* xrow = xbase + (size_t)tt * IDIM;
#pragma unroll
    for (int kc = 0; kc < IDIM / 32; ++kc) {
      const int kb = kc * 32;
      union { uint4 q[2]; v16bf v; } A;
      const float4* p0 = (const float4*)(xrow + kb + hi * 8);
      const float4* p1 = (const float4*)(xrow + kb + 16 + hi * 8);
      float4 a0 = p0[0], a1 = p0[1], b0 = p1[0], b1 = p1[1];
      A.q[0] = make_uint4(pk2(a0.x, a0.y), pk2(a0.z, a0.w), pk2(a1.x, a1.y), pk2(a1.z, a1.w));
      A.q[1] = make_uint4(pk2(b0.x, b0.y), pk2(b0.z, b0.w), pk2(b1.x, b1.y), pk2(b1.z, b1.w));
      const unsigned short* bp = ldsb + kb;
      v16bf B0 = ld_frag_lds(bp);
      v16bf B1 = ld_frag_lds(bp + 16 * LDS_STRIDE);
      v16bf B2 = ld_frag_lds(bp + 32 * LDS_STRIDE);
      v16bf B3 = ld_frag_lds(bp + 48 * LDS_STRIDE);
      const v8f c0v = (kc == 0) ? vbias[0] : acc[0];
      const v8f c1v = (kc == 0) ? vbias[1] : acc[1];
      const v8f c2v = (kc == 0) ? vbias[2] : acc[2];
      const v8f c3v = (kc == 0) ? vbias[3] : acc[3];
      acc[0] = __builtin_amdgcn_wmma_f32_16x16x32_bf16(false, A.v, false, B0, (short)0, c0v, false, false);
      acc[1] = __builtin_amdgcn_wmma_f32_16x16x32_bf16(false, A.v, false, B1, (short)0, c1v, false, false);
      acc[2] = __builtin_amdgcn_wmma_f32_16x16x32_bf16(false, A.v, false, B2, (short)0, c2v, false, false);
      acc[3] = __builtin_amdgcn_wmma_f32_16x16x32_bf16(false, A.v, false, B3, (short)0, c3v, false, false);
    }
  };

  // one LSTM timestep: consumes acc (x-partial+bias for step t), produces accnext for t+1
  auto do_step = [&](int t, v8f* acc, v8f* accnext) {
    const unsigned short* hin  = hbuf + (size_t)(t & 1) * (BATCH * HDIM);
    unsigned short*       hout = hbuf + (size_t)((t + 1) & 1) * (BATCH * HDIM);
    const unsigned short* hrow = hin + arow * HDIM;

    // warm the cache for the next timestep's x row while WMMAs run
    if (t + 1 < TSTEPS) {
      __builtin_prefetch(xbase + (size_t)(t + 1) * IDIM, 0, 0);
      __builtin_prefetch(xbase + (size_t)(t + 1) * IDIM + 16, 0, 0);
    }

    // ---- recurrent part: K = HDIM over bf16 h state (critical path) ----
#pragma unroll 4
    for (int kc = 0; kc < HDIM / 32; ++kc) {
      const int kb = kc * 32;                // k offset within h
      union { uint4 q[2]; v16bf v; } A;
      A.q[0] = *(const uint4*)(hrow + kb + hi * 8);
      A.q[1] = *(const uint4*)(hrow + kb + 16 + hi * 8);
      const unsigned short* bp = ldsb + IDIM + kb;
      v16bf B0 = ld_frag_lds(bp);
      v16bf B1 = ld_frag_lds(bp + 16 * LDS_STRIDE);
      v16bf B2 = ld_frag_lds(bp + 32 * LDS_STRIDE);
      v16bf B3 = ld_frag_lds(bp + 48 * LDS_STRIDE);
      acc[0] = __builtin_amdgcn_wmma_f32_16x16x32_bf16(false, A.v, false, B0, (short)0, acc[0], false, false);
      acc[1] = __builtin_amdgcn_wmma_f32_16x16x32_bf16(false, A.v, false, B1, (short)0, acc[1], false, false);
      acc[2] = __builtin_amdgcn_wmma_f32_16x16x32_bf16(false, A.v, false, B2, (short)0, acc[2], false, false);
      acc[3] = __builtin_amdgcn_wmma_f32_16x16x32_bf16(false, A.v, false, B3, (short)0, acc[3], false, false);
    }

    // ---- fused cell update: i,f,g,o -> c,h (bias already in acc) ----
    float hvals[8];
#pragma unroll
    for (int r = 0; r < 8; ++r) {
      float c  = fsig(acc[1][r]) * creg[r] + fsig(acc[0][r]) * ftanh(acc[2][r]);
      creg[r]  = c;
      float h  = fsig(acc[3][r]) * ftanh(c);
      hvals[r] = h;
      const int b = b_base + r;
      __builtin_nontemporal_store(h, &out[(size_t)BATCH * HDIM + ((size_t)b * TSTEPS + t) * HDIM + hg]);
      hout[b * HDIM + hg] = f2bf_hw(h);
    }
    if (t == TSTEPS - 1) {
#pragma unroll
      for (int r = 0; r < 8; ++r)
        __builtin_nontemporal_store(hvals[r], &out[(b_base + r) * HDIM + hg]);   // last_out
    }

    // ---- off-critical-path: next step's x-partial before arriving at barrier ----
    if (t + 1 < TSTEPS) compute_x_part(t + 1, accnext);

    // ---- grid barrier: all WGs must finish step t before anyone reads h(t+1) ----
    __threadfence();
    __syncthreads();
    if (threadIdx.x == 0) {
      __hip_atomic_fetch_add(counter, 1u, __ATOMIC_RELEASE, __HIP_MEMORY_SCOPE_AGENT);
      const unsigned want = (unsigned)(t + 1) * (unsigned)gridDim.x;
      while (__hip_atomic_load(counter, __ATOMIC_ACQUIRE, __HIP_MEMORY_SCOPE_AGENT) < want)
        __builtin_amdgcn_s_sleep(1);
    }
    __syncthreads();
    __threadfence();
  };

  v8f accA[4], accB[4];
  compute_x_part(0, accA);                   // t=0 x-partial (+bias) before entering the recurrence

  for (int t = 0; t < TSTEPS; t += 2) {      // ping-pong accumulators: no per-step copies
    do_step(t,     accA, accB);
    do_step(t + 1, accB, accA);
  }
}

extern "C" void kernel_launch(void* const* d_in, const int* in_sizes, int n_in,
                              void* d_out, int out_size, void* d_ws, size_t ws_size,
                              hipStream_t stream) {
  const float* x   = (const float*)d_in[0];
  const float* Wih = (const float*)d_in[1];
  const float* Whh = (const float*)d_in[2];
  const float* bih = (const float*)d_in[3];
  const float* bhh = (const float*)d_in[4];
  const float* h0  = (const float*)d_in[5];
  const float* c0  = (const float*)d_in[6];
  float* out = (float*)d_out;

  unsigned short* hbuf = (unsigned short*)d_ws;                       // 2 * 128*512 bf16 = 256 KB
  unsigned* counter = (unsigned*)((char*)d_ws + 2u * BATCH * HDIM * sizeof(unsigned short));

  lstm_prep<<<(BATCH * HDIM + 255) / 256, 256, 0, stream>>>(h0, hbuf, counter);

  const size_t ldsBytes = 64 * LDS_STRIDE * sizeof(unsigned short);   // ~81 KB (gfx1250: 320 KB/WGP)
  lstm_persistent<<<NWG, THREADS, ldsBytes, stream>>>(
      x, Wih, Whh, bih, bhh, c0, hbuf, counter, out);
}
